// QuantGRU_58909771432864
// MI455X (gfx1250) — 8-bit matrix operand order repaired and checked by reading, compiled, not run on hardware
//
#include <hip/hip_runtime.h>
#include <stdint.h>


#define T_STEPS 512
#define BATCH   64
#define IN_DIM  256
#define HID     256
#define H3      768

typedef int v8i __attribute__((ext_vector_type(8)));
typedef int v4i __attribute__((ext_vector_type(4)));

__device__ __forceinline__ int clampi(float x) {
    float r = rintf(x);
    r = fminf(fmaxf(r, -128.f), 127.f);
    return (int)r;
}
__device__ __forceinline__ int rne_shr(int x, int n) {
    return (x + (1 << (n - 1)) - 1 + ((x >> n) & 1)) >> n;
}
__device__ __forceinline__ int clamp8(int x) {
    return x < -128 ? -128 : (x > 127 ? 127 : x);
}

__device__ __forceinline__ v8i load_a_frag(const int8_t* row, int kt, int half) {
    v8i a;
#pragma unroll
    for (int i = 0; i < 4; ++i) {
        const int* p = (const int*)(row + kt * 64 + half * 8 + i * 16);
        a[2 * i]     = p[0];
        a[2 * i + 1] = p[1];
    }
    return a;
}
__device__ __forceinline__ v8i load_b_frag(const int8_t* row, int kt, int half) {
    v8i b;
#pragma unroll
    for (int i = 0; i < 4; ++i) {
        const int* p = (const int*)(row + kt * 64 + half * 8 + i * 16);
        b[2 * i]     = p[0];
        b[2 * i + 1] = p[1];
    }
    return b;
}

__global__ __launch_bounds__(256) void quant_weights(
    const float* __restrict__ W, const float* __restrict__ R,
    const float* __restrict__ bx, const float* __restrict__ br,
    int8_t* __restrict__ wqT, int8_t* __restrict__ rqT,
    short* __restrict__ bxq, short* __restrict__ brq) {
    int gid = blockIdx.x * 256 + threadIdx.x;
    if (gid < IN_DIM * H3) {
        int j = gid >> 8, i = gid & 255;
        wqT[gid] = (int8_t)clampi(W[(size_t)i * H3 + j] * 256.f);
    } else {
        int g2 = gid - IN_DIM * H3;
        int j = g2 >> 8, i = g2 & 255;
        rqT[g2] = (int8_t)clampi(R[(size_t)i * H3 + j] * 256.f);
    }
    if (gid < H3) {
        bxq[gid] = (short)clampi(bx[gid] * 256.f);
        brq[gid] = (short)clampi(br[gid] * 256.f);
    }
}

__global__ __launch_bounds__(256) void quant_x(
    const float* __restrict__ x, int8_t* __restrict__ xq) {
    size_t gid = (size_t)blockIdx.x * 256 + threadIdx.x;
    xq[gid] = (int8_t)clampi(x[gid] * 128.f);
}

__global__ __launch_bounds__(256) void gemm_wx(
    const int8_t* __restrict__ xq, const int8_t* __restrict__ wqT,
    int8_t* __restrict__ wxq) {
    int lane = threadIdx.x & 31, w = threadIdx.x >> 5;
    int half = lane >> 4, n16 = lane & 15;
    int mTile = blockIdx.x * 8 + w;

    const int8_t* arow = xq + (size_t)(mTile * 16 + n16) * IN_DIM;
    v8i a[4];
#pragma unroll
    for (int kt = 0; kt < 4; ++kt) a[kt] = load_a_frag(arow, kt, half);

    for (int nT = 0; nT < 48; nT += 2) {
        const int8_t* browA = wqT + (size_t)(nT * 16 + n16) * IN_DIM;
        const int8_t* browB = browA + 16 * IN_DIM;
        v8i accA = {0, 0, 0, 0, 0, 0, 0, 0};
        v8i accB = {0, 0, 0, 0, 0, 0, 0, 0};
#pragma unroll
        for (int kt = 0; kt < 4; ++kt) {
            v8i ba = load_b_frag(browA, kt, half);
            v8i bb = load_b_frag(browB, kt, half);
            accA = __builtin_amdgcn_wmma_i32_16x16x64_iu8(true, a[kt], true, ba,
                                                          accA, false, false);
            accB = __builtin_amdgcn_wmma_i32_16x16x64_iu8(true, a[kt], true, bb,
                                                          accB, false, false);
        }
        int8_t* out = wxq + (size_t)(mTile * 16 + half * 8) * H3 + nT * 16 + n16;
#pragma unroll
        for (int r = 0; r < 8; ++r) {
            out[(size_t)r * H3]      = (int8_t)clamp8(rne_shr(accA[r], 8));
            out[(size_t)r * H3 + 16] = (int8_t)clamp8(rne_shr(accB[r], 8));
        }
    }
}

#define SM_RQ   0
#define SM_V    196608
#define SM_H    245760
#define SM_HQ   278528
#define SM_BX   294912
#define SM_BR   296448
#define SM_SIZE 297984

__global__ __launch_bounds__(1024, 1) void gru_scan(
    const int8_t* __restrict__ wxq, const int8_t* __restrict__ rqT,
    const short* __restrict__ bxq, const short* __restrict__ brq,
    const float* __restrict__ h0, float* __restrict__ hs) {
    extern __shared__ char smem[];
    int8_t* rq_s = (int8_t*)(smem + SM_RQ);
    int8_t* v_s  = (int8_t*)(smem + SM_V);
    short*  h_s  = (short*)(smem + SM_H);
    int8_t* hq_s = (int8_t*)(smem + SM_HQ);
    short*  bx_s = (short*)(smem + SM_BX);
    short*  br_s = (short*)(smem + SM_BR);

    const int tid = threadIdx.x;
    {
        const int4* src = (const int4*)rqT;
        int4* dst = (int4*)rq_s;
        for (int i = tid; i < (HID * H3) / 16; i += 1024) dst[i] = src[i];
    }
    for (int e = tid; e < BATCH * HID; e += 1024) {
        float r = rintf(h0[e] * 128.f);
        h_s[e]  = (short)(int)fminf(fmaxf(r, -32768.f), 32767.f);
        hq_s[e] = (int8_t)clampi(h0[e] * 128.f);
    }
    for (int j = tid; j < H3; j += 1024) { bx_s[j] = bxq[j]; br_s[j] = brq[j]; }
    __syncthreads();

    const int lane = tid & 31, w = tid >> 5;
    const int half = lane >> 4, n16 = lane & 15;
    const int mT   = w & 3;
    const int nT0  = (w >> 2) * 6;
    const int8_t* arow = hq_s + (mT * 16 + n16) * HID;

    for (int t = 0; t < T_STEPS; ++t) {
        v8i a[4];
#pragma unroll
        for (int kt = 0; kt < 4; ++kt) a[kt] = load_a_frag(arow, kt, half);
#pragma unroll
        for (int j = 0; j < 6; j += 2) {
            const int8_t* browA = rq_s + ((nT0 + j) * 16 + n16) * HID;
            const int8_t* browB = browA + 16 * HID;
            v8i accA = {0, 0, 0, 0, 0, 0, 0, 0};
            v8i accB = {0, 0, 0, 0, 0, 0, 0, 0};
#pragma unroll
            for (int kt = 0; kt < 4; ++kt) {
                v8i ba = load_b_frag(browA, kt, half);
                v8i bb = load_b_frag(browB, kt, half);
                accA = __builtin_amdgcn_wmma_i32_16x16x64_iu8(true, a[kt], true, ba,
                                                              accA, false, false);
                accB = __builtin_amdgcn_wmma_i32_16x16x64_iu8(true, a[kt], true, bb,
                                                              accB, false, false);
            }
            int colA = (nT0 + j) * 16 + n16;
            int brA = br_s[colA], brB = br_s[colA + 16];
            int8_t* vout = v_s + (mT * 16 + half * 8) * H3 + colA;
#pragma unroll
            for (int r = 0; r < 8; ++r) {
                int rhA = clamp8(rne_shr(accA[r], 8));
                int rhB = clamp8(rne_shr(accB[r], 8));
                vout[r * H3]      = (int8_t)clamp8(rne_shr(2 * rhA + brA, 1));
                vout[r * H3 + 16] = (int8_t)clamp8(rne_shr(2 * rhB + brB, 1));
            }
        }
        __syncthreads();

        const int8_t* wrow_base = wxq + (size_t)t * (BATCH * H3);
        if (t + 1 < T_STEPS)
            __builtin_prefetch(wrow_base + BATCH * H3 + tid * 48, 0, 1);
#pragma unroll
        for (int g = 0; g < 4; ++g) {
            int idx = g * 4096 + tid * 4;
            int b = idx >> 8, jj = idx & 255;
            const int8_t* wrow = wrow_base + b * H3;
            int wzp = *(const int*)(wrow + jj);
            int wrp = *(const int*)(wrow + jj + 256);
            int wgp = *(const int*)(wrow + jj + 512);
            const int8_t* vrow = v_s + b * H3;
            int vzp = *(const int*)(vrow + jj);
            int vrp = *(const int*)(vrow + jj + 256);
            int vgp = *(const int*)(vrow + jj + 512);

            float4 outv;
            int hq_pack = 0;
            int hn4[4];
#pragma unroll
            for (int k = 0; k < 4; ++k) {
                int wz = (int)(int8_t)(wzp >> (8 * k));
                int wr = (int)(int8_t)(wrp >> (8 * k));
                int wg = (int)(int8_t)(wgp >> (8 * k));
                int vz = (int)(int8_t)(vzp >> (8 * k));
                int vr = (int)(int8_t)(vrp >> (8 * k));
                int vg = (int)(int8_t)(vgp >> (8 * k));
                int bz  = bx_s[jj + k];
                int brg = bx_s[jj + k + 256];
                int bg  = bx_s[jj + k + 512];

                int zq = clamp8(rne_shr(2 * wz + bz + 2 * vz, 1));
                int zi = clampi(128.f / (1.f + __expf(-(float)zq * 0.0078125f)));
                int rq = clamp8(rne_shr(2 * wr + brg + 2 * vr, 1));
                int ri = clampi(128.f / (1.f + __expf(-(float)rq * 0.0078125f)));
                int rr = clamp8(rne_shr(ri * vg, 7));
                int gq = clamp8(rne_shr(2 * wg + bg + 2 * rr, 1));
                int gi = clampi(128.f * tanhf((float)gq * 0.0078125f));

                int hold = (int)h_s[idx + k];
                int oi = clamp8(rne_shr(zi * hold, 7));
                int ni = clamp8(rne_shr((128 - zi) * gi, 7));
                int hn = oi + ni;
                hn4[k] = hn;
                hq_pack |= (clamp8(hn) & 255) << (8 * k);
                (&outv.x)[k] = (float)hn * 0.0078125f;
            }
            *(int*)(hq_s + idx) = hq_pack;
            int* hp = (int*)(h_s + idx);
            hp[0] = (hn4[0] & 0xffff) | (hn4[1] << 16);
            hp[1] = (hn4[2] & 0xffff) | (hn4[3] << 16);
            *(float4*)(hs + (size_t)t * (BATCH * HID) + idx) = outv;
        }
        __syncthreads();
    }
}

extern "C" void kernel_launch(void* const* d_in, const int* in_sizes, int n_in,
                              void* d_out, int out_size, void* d_ws, size_t ws_size,
                              hipStream_t stream) {
    const float* x  = (const float*)d_in[0];
    const float* h0 = (const float*)d_in[1];
    const float* W  = (const float*)d_in[2];
    const float* R  = (const float*)d_in[3];
    const float* bx = (const float*)d_in[4];
    const float* br = (const float*)d_in[5];
    char* ws = (char*)d_ws;
    int8_t* wqT = (int8_t*)(ws);
    int8_t* rqT = (int8_t*)(ws + 196608);
    short*  bxq = (short*)(ws + 393216);
    short*  brq = (short*)(ws + 394752);
    int8_t* xq  = (int8_t*)(ws + 396288);
    int8_t* wxq = (int8_t*)(ws + 8784896);
    float*  hs  = (float*)d_out;

    quant_weights<<<1536, 256, 0, stream>>>(W, R, bx, br, wqT, rqT, bxq, brq);
    quant_x<<<(T_STEPS * BATCH * IN_DIM) / 256, 256, 0, stream>>>(x, xq);
    gemm_wx<<<256, 256, 0, stream>>>(xq, wqT, wxq);
    gru_scan<<<1, 1024, SM_SIZE, stream>>>(wxq, rqT, bxq, brq, h0, hs);
}
